// MPNN_8650064134367
// MI455X (gfx1250) — matrix operand order repaired and checked by reading, not compiled, not run on hardware
//
#include <hip/hip_runtime.h>
#include <hip/hip_bf16.h>

// ---------------------------------------------------------------------------
// MPNN (lin0 -> 3x{lin1,GRU} -> Set2Set -> head) for MI455X / gfx1250.
// Big GEMMs (N=1e6 rows, K=64) on v_wmma_f32_16x16x32_f16 (f32 accumulate).
// ~182 GFLOP / ~3.3 GB HBM -> memory bound at 23.3 TB/s; WMMA keeps the
// math off the critical path.  Node tiles are DMA'd into LDS with
// global_load_async_to_lds_b128 (ASYNCcnt) overlapped with weight staging.
// Fragments load as ds_load_b128 pairs (contiguous 16B runs per lane).
// ---------------------------------------------------------------------------

typedef __attribute__((ext_vector_type(16))) _Float16 v16h;
typedef __attribute__((ext_vector_type(8)))  float    v8f;
typedef __attribute__((ext_vector_type(4)))  unsigned uint4v;

#define WAVES 8   // 256-thread blocks = 8 wave32

__device__ __forceinline__ v8f wmma16(v16h a, v16h b, v8f c) {
  // D = A(16x32 f16) x B(32x16 f16) + C(16x16 f32)
  return __builtin_amdgcn_wmma_f32_16x16x32_f16(
      /*neg_a=*/false, a, /*neg_b=*/false, b,
      /*c_mod=*/(short)0, c, /*reuse_a=*/false, /*reuse_b=*/false);
}

// A fragment: 16x32 f16 tile, row-major, row stride 64 halfs.
// ISA layout -> per lane, two contiguous 16B runs:
//   halfs [m*64+kbase+hi .. +7] and [m*64+kbase+16+hi .. +7]  (hi = (lane>=16)*8)
__device__ __forceinline__ v16h load_A_frag(const _Float16* t16, int lane, int kbase) {
  union { v16h v; uint4v q[2]; } f;
  const int m  = lane & 15;
  const int hi = (lane >> 4) * 8;
  const _Float16* p = t16 + m * 64 + kbase + hi;
  f.q[0] = *(const uint4v*)p;          // ds_load_b128
  f.q[1] = *(const uint4v*)(p + 16);   // ds_load_b128
  return f.v;
}

// B fragment for B = W^T: lane n reads weight row (otile*16 + n) as two 8-half
// runs: K=kbase+hi..+7 and K=kbase+16+hi..+7 (hi = (lane>=16)*8).
__device__ __forceinline__ v16h load_B_frag(const _Float16* W, int lane, int otile, int kbase) {
  union { v16h v; uint4v q[2]; } f;
  const int col = lane & 15;
  const int ks  = (lane >> 4) * 8;
  const _Float16* p = W + (size_t)(otile * 16 + col) * 64 + kbase + ks;
  f.q[0] = *(const uint4v*)p;          // ds_load_b128
  f.q[1] = *(const uint4v*)(p + 16);   // ds_load_b128
  return f.v;
}

__device__ __forceinline__ float sigmoidf(float x) { return 1.f / (1.f + __expf(-x)); }

union HF2 { _Float16 h[2]; unsigned u; };

__device__ __forceinline__ unsigned pack_f16x2(float a, float b) {
  HF2 p; p.h[0] = (_Float16)a; p.h[1] = (_Float16)b; return p.u;
}

// ---------------------------------------------------------------------------
// Kernel 1: h = relu(x @ lin0_w^T + b).  One wave per 16-node tile.
// ---------------------------------------------------------------------------
__global__ __launch_bounds__(256) void lin0_kernel(
    const float* __restrict__ x, float* __restrict__ h,
    const float* __restrict__ W, const float* __restrict__ b, int n_tiles) {
  extern __shared__ char smem[];
  _Float16* sW  = (_Float16*)smem;                  // 64*64 halfs
  float*    sB  = (float*)(sW + 64 * 64);           // 64 floats
  _Float16* sX  = (_Float16*)(sB + 64);             // WAVES * 16*64 halfs

  const int tid = threadIdx.x;
  {
    const float2* Wv = (const float2*)W;
    unsigned* sWu = (unsigned*)sW;
    for (int i = tid; i < 64 * 64 / 2; i += 256) {
      float2 v = Wv[i];
      sWu[i] = pack_f16x2(v.x, v.y);
    }
  }
  if (tid < 64) sB[tid] = b[tid];
  __syncthreads();

  const int wave = tid >> 5, lane = tid & 31;
  const int tile = blockIdx.x * WAVES + wave;
  if (tile < n_tiles) {
    const size_t base = (size_t)tile * 1024;        // 16 rows * 64 cols
    _Float16* x16 = sX + wave * 1024;
    const float4* src = (const float4*)(x + base);
    unsigned* x16u = (unsigned*)x16;
#pragma unroll
    for (int it = 0; it < 8; ++it) {
      const int idx = it * 32 + lane;               // float4 index (0..255)
      float4 v = src[idx];
      x16u[idx * 2 + 0] = pack_f16x2(v.x, v.y);
      x16u[idx * 2 + 1] = pack_f16x2(v.z, v.w);
    }
    v16h a0 = load_A_frag(x16, lane, 0);
    v16h a1 = load_A_frag(x16, lane, 32);
    const int ncol = lane & 15, mrow0 = (lane >> 4) * 8;
#pragma unroll
    for (int t = 0; t < 4; ++t) {
      v8f acc = {};
      acc = wmma16(a0, load_B_frag(sW, lane, t, 0),  acc);
      acc = wmma16(a1, load_B_frag(sW, lane, t, 32), acc);
      const float bb = sB[t * 16 + ncol];
#pragma unroll
      for (int r = 0; r < 8; ++r) {
        float v = acc[r] + bb;
        h[base + (size_t)(mrow0 + r) * 64 + t * 16 + ncol] = v > 0.f ? v : 0.f;
      }
    }
  }
}

// ---------------------------------------------------------------------------
// Kernel 2: fused {m = relu(h@W1^T+b1); h = GRU(m,h)} in place.
// Node tile is DMA'd global->LDS via the async engine, overlapped with the
// cooperative weight staging; 56 WMMAs per wave.  In-place safe: a wave reads
// its whole tile before writing any of it; tiles are disjoint across waves.
// ---------------------------------------------------------------------------
__global__ __launch_bounds__(256) void gru_fused_kernel(
    float* __restrict__ h_g,
    const float* __restrict__ W1,  const float* __restrict__ b1,
    const float* __restrict__ Wih, const float* __restrict__ Whh,
    const float* __restrict__ bih, const float* __restrict__ bhh, int n_tiles) {
  extern __shared__ char smem[];
  _Float16* sW1  = (_Float16*)smem;                 // 64*64
  _Float16* sWih = sW1 + 64 * 64;                   // 192*64
  _Float16* sWhh = sWih + 192 * 64;                 // 192*64
  float* sB1  = (float*)(sWhh + 192 * 64);          // 64
  float* sBih = sB1 + 64;                           // 192
  float* sBhh = sBih + 192;                         // 192
  float*    sHf  = sBhh + 192;                      // WAVES*1024 f32 (h tile)
  _Float16* sH16 = (_Float16*)(sHf + WAVES * 1024); // WAVES*1024 f16
  _Float16* sM16 = sH16 + WAVES * 1024;             // WAVES*1024 f16

  const int tid = threadIdx.x;
  const int wave = tid >> 5, lane = tid & 31;
  const int tile = blockIdx.x * WAVES + wave;
  float* hf = sHf + wave * 1024;

  // --- Async DMA of this wave's 4KB node tile into its private LDS slot ---
  // (issued before weight staging so the TDM-side transfer overlaps it).
  if (tile < n_tiles) {
    const unsigned lds_dst0 = (unsigned)(uintptr_t)hf;          // LDS = addr[31:0]
    const unsigned long long g_src0 =
        (unsigned long long)(uintptr_t)(h_g + (size_t)tile * 1024);
#pragma unroll
    for (int it = 0; it < 8; ++it) {
      const unsigned           dst = lds_dst0 + (unsigned)((it * 32 + lane) * 16);
      const unsigned long long src = g_src0 + (unsigned long long)((it * 32 + lane) * 16);
      asm volatile("global_load_async_to_lds_b128 %0, %1, off"
                   :: "v"(dst), "v"(src) : "memory");
    }
  }

  // --- Cooperative weight staging (f32 -> f16, packed ds_store_b32) ---
  {
    const float2* W1v  = (const float2*)W1;
    const float2* Wihv = (const float2*)Wih;
    const float2* Whhv = (const float2*)Whh;
    unsigned* sW1u  = (unsigned*)sW1;
    unsigned* sWihu = (unsigned*)sWih;
    unsigned* sWhhu = (unsigned*)sWhh;
    for (int i = tid; i < 64 * 64 / 2; i += 256) {
      float2 v = W1v[i];  sW1u[i]  = pack_f16x2(v.x, v.y);
    }
    for (int i = tid; i < 192 * 64 / 2; i += 256) {
      float2 v = Wihv[i]; sWihu[i] = pack_f16x2(v.x, v.y);
      float2 w = Whhv[i]; sWhhu[i] = pack_f16x2(w.x, w.y);
    }
    if (tid < 64)  sB1[tid]  = b1[tid];
    if (tid < 192) { sBih[tid] = bih[tid]; sBhh[tid] = bhh[tid]; }
  }
  __syncthreads();

  if (tile < n_tiles) {
    const size_t base = (size_t)tile * 1024;
    _Float16* h16 = sH16 + wave * 1024;
    _Float16* m16 = sM16 + wave * 1024;
    unsigned* h16u = (unsigned*)h16;

    // Wait for the async tile DMA, then build the f16 copy from LDS.
    asm volatile("s_wait_asynccnt 0x0" ::: "memory");
#pragma unroll
    for (int it = 0; it < 8; ++it) {
      const int idx = it * 32 + lane;
      float4 v = ((const float4*)hf)[idx];          // ds_load_b128
      h16u[idx * 2 + 0] = pack_f16x2(v.x, v.y);
      h16u[idx * 2 + 1] = pack_f16x2(v.z, v.w);
    }
    v16h hA0 = load_A_frag(h16, lane, 0);
    v16h hA1 = load_A_frag(h16, lane, 32);

    const int ncol = lane & 15, mrow0 = (lane >> 4) * 8;

    // m = relu(h @ W1^T + b1) -> staged to LDS for A-refragmentation.
#pragma unroll
    for (int t = 0; t < 4; ++t) {
      v8f acc = {};
      acc = wmma16(hA0, load_B_frag(sW1, lane, t, 0),  acc);
      acc = wmma16(hA1, load_B_frag(sW1, lane, t, 32), acc);
      const float bb = sB1[t * 16 + ncol];
#pragma unroll
      for (int r = 0; r < 8; ++r) {
        float v = acc[r] + bb;
        m16[(mrow0 + r) * 64 + t * 16 + ncol] = (_Float16)(v > 0.f ? v : 0.f);
      }
    }
    v16h mA0 = load_A_frag(m16, lane, 0);
    v16h mA1 = load_A_frag(m16, lane, 32);

    // Gates: gi = m@Wih^T, gh = h@Whh^T (192 outputs each; r|z|n chunks of 64).
#pragma unroll
    for (int ct = 0; ct < 4; ++ct) {
      v8f ir = {}, iz = {}, inn = {}, hr = {}, hz = {}, hn = {};
      ir  = wmma16(mA0, load_B_frag(sWih, lane, ct,     0),  ir);
      ir  = wmma16(mA1, load_B_frag(sWih, lane, ct,     32), ir);
      iz  = wmma16(mA0, load_B_frag(sWih, lane, 4 + ct, 0),  iz);
      iz  = wmma16(mA1, load_B_frag(sWih, lane, 4 + ct, 32), iz);
      inn = wmma16(mA0, load_B_frag(sWih, lane, 8 + ct, 0),  inn);
      inn = wmma16(mA1, load_B_frag(sWih, lane, 8 + ct, 32), inn);
      hr  = wmma16(hA0, load_B_frag(sWhh, lane, ct,     0),  hr);
      hr  = wmma16(hA1, load_B_frag(sWhh, lane, ct,     32), hr);
      hz  = wmma16(hA0, load_B_frag(sWhh, lane, 4 + ct, 0),  hz);
      hz  = wmma16(hA1, load_B_frag(sWhh, lane, 4 + ct, 32), hz);
      hn  = wmma16(hA0, load_B_frag(sWhh, lane, 8 + ct, 0),  hn);
      hn  = wmma16(hA1, load_B_frag(sWhh, lane, 8 + ct, 32), hn);

      const int c = ct * 16 + ncol;
      const float bir = sBih[c], biz = sBih[64 + c], bin = sBih[128 + c];
      const float bhr = sBhh[c], bhz = sBhh[64 + c], bhn = sBhh[128 + c];
#pragma unroll
      for (int r = 0; r < 8; ++r) {
        const int mr = mrow0 + r;
        const float hval = hf[mr * 64 + c];
        const float rg = sigmoidf(ir[r] + bir + hr[r] + bhr);
        const float zg = sigmoidf(iz[r] + biz + hz[r] + bhz);
        const float ng = tanhf(inn[r] + bin + rg * (hn[r] + bhn));
        h_g[base + (size_t)mr * 64 + c] = (1.f - zg) * ng + zg * hval;
      }
    }
  }
}

// ---------------------------------------------------------------------------
// Set2Set helpers
// ---------------------------------------------------------------------------
__device__ __forceinline__ unsigned enc_f32(float f) {
  unsigned b = __float_as_uint(f);
  return (b & 0x80000000u) ? ~b : (b | 0x80000000u);
}
__device__ __forceinline__ float dec_f32(unsigned e) {
  unsigned b = (e & 0x80000000u) ? (e ^ 0x80000000u) : ~e;
  return __uint_as_float(b);
}

__global__ void fill_f32_kernel(float* p, float v, long n) {
  long i = (long)blockIdx.x * blockDim.x + threadIdx.x;
  for (; i < n; i += (long)gridDim.x * blockDim.x) p[i] = v;
}
__global__ void fill_u32_kernel(unsigned* p, unsigned v, long n) {
  long i = (long)blockIdx.x * blockDim.x + threadIdx.x;
  for (; i < n; i += (long)gridDim.x * blockDim.x) p[i] = v;
}

// LSTM cell on [B,128] q_star: one 64-thread block per graph.
__global__ __launch_bounds__(64) void lstm_kernel(
    const float* __restrict__ q_star, float* __restrict__ hl, float* __restrict__ cl,
    const float* __restrict__ Wih, const float* __restrict__ Whh,
    const float* __restrict__ bih, const float* __restrict__ bhh) {
  __shared__ float sq[128];
  __shared__ float sh[64];
  const int g = blockIdx.x, c = threadIdx.x;
  sq[c] = q_star[(size_t)g * 128 + c];
  sq[c + 64] = q_star[(size_t)g * 128 + 64 + c];
  sh[c] = hl[(size_t)g * 64 + c];
  __syncthreads();
  float acc[4];
#pragma unroll
  for (int gate = 0; gate < 4; ++gate) {
    const int row = gate * 64 + c;
    float a = bih[row] + bhh[row];
    const float* wi = Wih + (size_t)row * 128;
    for (int k = 0; k < 128; ++k) a += sq[k] * wi[k];
    const float* wh = Whh + (size_t)row * 64;
    for (int k = 0; k < 64; ++k) a += sh[k] * wh[k];
    acc[gate] = a;
  }
  const float ig = sigmoidf(acc[0]), fg = sigmoidf(acc[1]);
  const float gg = tanhf(acc[2]),    og = sigmoidf(acc[3]);
  const float cn = fg * cl[(size_t)g * 64 + c] + ig * gg;
  cl[(size_t)g * 64 + c] = cn;
  hl[(size_t)g * 64 + c] = og * tanhf(cn);
}

// e[i] = dot64(h[i], q[batch[i]]); segment max via ordered-uint atomicMax.
__global__ __launch_bounds__(256) void attn_e_kernel(
    const float* __restrict__ h, const float* __restrict__ q,
    const int* __restrict__ batch, float* __restrict__ e,
    unsigned* __restrict__ gmax, int n_nodes) {
  const int lane = threadIdx.x & 31;
  int wv = blockIdx.x * (blockDim.x >> 5) + (threadIdx.x >> 5);
  const int nwaves = gridDim.x * (blockDim.x >> 5);
  for (int i = wv; i < n_nodes; i += nwaves) {
    const int g = batch[i];
    float s = h[(size_t)i * 64 + lane]      * q[(size_t)g * 64 + lane]
            + h[(size_t)i * 64 + 32 + lane] * q[(size_t)g * 64 + 32 + lane];
#pragma unroll
    for (int off = 16; off; off >>= 1) s += __shfl_xor(s, off, 32);
    if (lane == 0) {
      e[i] = s;
      atomicMax(&gmax[g], enc_f32(s));
    }
  }
}

// ex = exp(e - max[batch]); segment sum.
__global__ void attn_expsum_kernel(
    const float* __restrict__ e, const int* __restrict__ batch,
    const unsigned* __restrict__ gmax, float* __restrict__ ex,
    float* __restrict__ gsum, int n_nodes) {
  const int i = blockIdx.x * blockDim.x + threadIdx.x;
  if (i >= n_nodes) return;
  const int g = batch[i];
  const float v = __expf(e[i] - dec_f32(gmax[g]));
  ex[i] = v;
  atomicAdd(&gsum[g], v);
}

// r[g] += a_i * h[i]; batch is sorted, so accumulate runs before atomics.
__global__ __launch_bounds__(64) void attn_wsum_kernel(
    const float* __restrict__ h, const float* __restrict__ ex,
    const float* __restrict__ gsum, const int* __restrict__ batch,
    float* __restrict__ r, int n_nodes) {
  const int NB = 128;
  const int c = threadIdx.x;
  const int start = blockIdx.x * NB;
  if (start >= n_nodes) return;
  const int end = (start + NB < n_nodes) ? start + NB : n_nodes;
  int cur = batch[start];
  float acc = 0.f;
  for (int i = start; i < end; ++i) {
    const int g = batch[i];
    if (g != cur) {
      atomicAdd(&r[(size_t)cur * 64 + c], acc);
      acc = 0.f; cur = g;
    }
    const float a = ex[i] / (gsum[g] + 1e-16f);
    acc += a * h[(size_t)i * 64 + c];
  }
  atomicAdd(&r[(size_t)cur * 64 + c], acc);
}

__global__ void pack_qstar_kernel(const float* __restrict__ hl,
                                  const float* __restrict__ r,
                                  float* __restrict__ q_star, int B) {
  const int idx = blockIdx.x * blockDim.x + threadIdx.x;
  if (idx >= B * 128) return;
  const int g = idx >> 7, j = idx & 127;
  q_star[idx] = (j < 64) ? hl[(size_t)g * 64 + j] : r[(size_t)g * 64 + (j - 64)];
}

// Head: y = relu(q_star@W2^T+b2) @ W3^T + b3.  One 64-thread block per graph.
__global__ __launch_bounds__(64) void head_kernel(
    const float* __restrict__ q_star,
    const float* __restrict__ W2, const float* __restrict__ b2,
    const float* __restrict__ W3, const float* __restrict__ b3,
    float* __restrict__ y) {
  __shared__ float sq[128];
  __shared__ float sh[64];
  const int g = blockIdx.x, c = threadIdx.x;
  sq[c] = q_star[(size_t)g * 128 + c];
  sq[c + 64] = q_star[(size_t)g * 128 + 64 + c];
  __syncthreads();
  float a = b2[c];
  const float* w = W2 + (size_t)c * 128;
  for (int k = 0; k < 128; ++k) a += sq[k] * w[k];
  a = a > 0.f ? a : 0.f;
  sh[c] = a * W3[c];
  __syncthreads();
  for (int off = 32; off; off >>= 1) {
    if (c < off) sh[c] += sh[c + off];
    __syncthreads();
  }
  if (c == 0) y[g] = sh[0] + b3[0];
}

// ---------------------------------------------------------------------------
// Host side
// ---------------------------------------------------------------------------
extern "C" void kernel_launch(void* const* d_in, const int* in_sizes, int n_in,
                              void* d_out, int out_size, void* d_ws, size_t ws_size,
                              hipStream_t stream) {
  const float* x      = (const float*)d_in[0];
  const int*   batch  = (const int*)d_in[1];
  const float* lin0_w = (const float*)d_in[2];
  const float* lin0_b = (const float*)d_in[3];
  const float* lin1_w = (const float*)d_in[4];
  const float* lin1_b = (const float*)d_in[5];
  const float* gwih   = (const float*)d_in[6];
  const float* gwhh   = (const float*)d_in[7];
  const float* gbih   = (const float*)d_in[8];
  const float* gbhh   = (const float*)d_in[9];
  const float* lwih   = (const float*)d_in[10];
  const float* lwhh   = (const float*)d_in[11];
  const float* lbih   = (const float*)d_in[12];
  const float* lbhh   = (const float*)d_in[13];
  const float* lin2_w = (const float*)d_in[14];
  const float* lin2_b = (const float*)d_in[15];
  const float* lin3_w = (const float*)d_in[16];
  const float* lin3_b = (const float*)d_in[17];
  float* y = (float*)d_out;

  const int N = in_sizes[0] / 64;   // 1,000,000 nodes
  const int B = out_size;           // 8192 graphs

  // Workspace carve-up.
  char* w = (char*)d_ws;
  float* h      = (float*)w; w += (size_t)N * 64 * 4;
  float* e      = (float*)w; w += (size_t)N * 4;
  float* ex     = (float*)w; w += (size_t)N * 4;
  unsigned* gmax= (unsigned*)w; w += (size_t)B * 4;
  float* gsum   = (float*)w; w += (size_t)B * 4;
  float* q_star = (float*)w; w += (size_t)B * 128 * 4;
  float* hl     = (float*)w; w += (size_t)B * 64 * 4;
  float* cl     = (float*)w; w += (size_t)B * 64 * 4;
  float* rbuf   = (float*)w; w += (size_t)B * 64 * 4;
  (void)ws_size; (void)n_in;

  const int n_tiles = N / 16;
  const int nblk    = (n_tiles + WAVES - 1) / WAVES;

  const size_t lds_lin0 = 64 * 64 * 2 + 64 * 4 + WAVES * 1024 * 2;
  const size_t lds_gru  = (64 + 192 + 192) * 64 * 2 + (64 + 192 + 192) * 4
                        + WAVES * 1024 * (4 + 2 + 2);

  // 1) h = relu(x @ lin0_w^T + b)
  lin0_kernel<<<nblk, 256, lds_lin0, stream>>>(x, h, lin0_w, lin0_b, n_tiles);

  // 2) 3 fused lin1+GRU passes, in place on h
  for (int it = 0; it < 3; ++it)
    gru_fused_kernel<<<nblk, 256, lds_gru, stream>>>(
        h, lin1_w, lin1_b, gwih, gwhh, gbih, gbhh, n_tiles);

  // 3) Set2Set, 3 processing steps
  fill_f32_kernel<<<256, 256, 0, stream>>>(q_star, 0.f, (long)B * 128);
  fill_f32_kernel<<<64, 256, 0, stream>>>(hl, 0.f, (long)B * 64);
  fill_f32_kernel<<<64, 256, 0, stream>>>(cl, 0.f, (long)B * 64);

  const unsigned ENC_NEG_INF = 0x007FFFFFu;  // enc_f32(-inf)
  for (int s = 0; s < 3; ++s) {
    lstm_kernel<<<B, 64, 0, stream>>>(q_star, hl, cl, lwih, lwhh, lbih, lbhh);

    fill_u32_kernel<<<32, 256, 0, stream>>>(gmax, ENC_NEG_INF, (long)B);
    fill_f32_kernel<<<32, 256, 0, stream>>>(gsum, 0.f, (long)B);
    fill_f32_kernel<<<64, 256, 0, stream>>>(rbuf, 0.f, (long)B * 64);

    attn_e_kernel<<<2048, 256, 0, stream>>>(h, hl, batch, e, gmax, N);
    attn_expsum_kernel<<<(N + 255) / 256, 256, 0, stream>>>(e, batch, gmax, ex, gsum, N);
    attn_wsum_kernel<<<(N + 127) / 128, 64, 0, stream>>>(h, ex, gsum, batch, rbuf, N);
    pack_qstar_kernel<<<(B * 128 + 255) / 256, 256, 0, stream>>>(hl, rbuf, q_star, B);
  }

  // 4) Head
  head_kernel<<<B, 64, 0, stream>>>(q_star, lin2_w, lin2_b, lin3_w, lin3_b, y);
}
